// Generator_73023033966837
// MI455X (gfx1250) — matrix operand order repaired and checked by reading, not compiled, not run on hardware
//
#include <hip/hip_runtime.h>
#include <math.h>

typedef __attribute__((ext_vector_type(16))) _Float16 v16h;
typedef __attribute__((ext_vector_type(8)))  float    v8f;
typedef __attribute__((ext_vector_type(4)))  float    v4f;   // clang vector: OK for nontemporal builtins

// Problem dims (fixed by the reference's setup_inputs)
constexpr int Bc = 8;
constexpr int Tc = 256;
constexpr int Sc = 512;
constexpr int Dc = 768;
constexpr int Hc = 8;
constexpr int Vc = 32000;
constexpr int VCHUNK = 8000;   // V processed in 4 LDS-resident chunks

// ---------------------------------------------------------------------------
// Kernel 1: p_gen = sigmoid(dec_output @ W + b) via v_wmma_f32_16x16x32_f16.
// One wave32 per block computes 16 rows. f32 inputs are split h + residual and
// three WMMAs per K-step recover ~f32 accuracy (hA*hB + hA*rB + rA*hB).
// ---------------------------------------------------------------------------
__global__ __launch_bounds__(32) void pgen_wmma_kernel(
    const float* __restrict__ dec,    // (B*T, D)
    const float* __restrict__ Wv,     // (D)
    const float* __restrict__ bptr,   // (1)
    float* __restrict__ pgen)         // (B*T)
{
  const int lane = threadIdx.x;           // 0..31
  const int hsel = lane >> 4;             // lane group
  const int mrow = lane & 15;             // A-matrix row for this lane
  const int rowA = blockIdx.x * 16 + mrow;

  v8f c = {};
  for (int kb = 0; kb < Dc; kb += 32) {
    v16h hA, rA, hB, rB;
#pragma unroll
    for (int j = 0; j < 16; ++j) {
      // A 16x32 f16 layout: K = j + (j>=8 ? 8 : 0) + 8*hsel
      int ka = kb + j + ((j >= 8) ? 8 : 0) + 8 * hsel;
      float x = dec[(size_t)rowA * Dc + ka];
      _Float16 hx = (_Float16)x;
      hA[j] = hx;
      rA[j] = (_Float16)(x - (float)hx);
      // B 32x16 f16 layout: K = j + (j>=8 ? 8 : 0) + 8*hsel, W broadcast across all N columns
      int kw = kb + j + ((j >= 8) ? 8 : 0) + 8 * hsel;
      float w = Wv[kw];
      _Float16 hw = (_Float16)w;
      hB[j] = hw;
      rB[j] = (_Float16)(w - (float)hw);
    }
    c = __builtin_amdgcn_wmma_f32_16x16x32_f16(false, hA, false, hB, (short)0, c, false, false);
    c = __builtin_amdgcn_wmma_f32_16x16x32_f16(false, hA, false, rB, (short)0, c, false, false);
    c = __builtin_amdgcn_wmma_f32_16x16x32_f16(false, rA, false, hB, (short)0, c, false, false);
  }

  // D layout: VGPR v, lanes 0-15 -> (M=v, N=lane); lanes 16-31 -> (M=v+8, N=lane-16).
  // Column N=0 lives in lane 0 (M=0..7) and lane 16 (M=8..15).
  if ((lane & 15) == 0) {
    const float bias = bptr[0];
    const int mbase = hsel * 8;
#pragma unroll
    for (int v = 0; v < 8; ++v) {
      float z = c[v] + bias;
      pgen[blockIdx.x * 16 + mbase + v] = 1.0f / (1.0f + expf(-z));
    }
  }
}

// ---------------------------------------------------------------------------
// Kernel 2: one workgroup per (b,t) row.
//   - mean over H + softmax over S (block reduce in LDS); NT loads (read-once)
//   - scatter (1-p)*att into LDS copy buffer (ds_add_f32 atomics), chunked
//   - online softmax stats over V: one 128-bit-vectorized global sweep
//   - emit log(p*softmax(x) + copy + 1e-3): b128 loads (L2-resident re-read)
//     and non-temporal b128 stores so the 262MB output stream doesn't evict
//     the final_output rows from L2.
// ---------------------------------------------------------------------------
__global__ __launch_bounds__(512) void fused_row_kernel(
    const float* __restrict__ finalo,  // (B*T, V)
    const float* __restrict__ attw,    // (B, H, T, S)
    const int*   __restrict__ enc,     // (B, S)
    const float* __restrict__ pgen,    // (B*T)
    float* __restrict__ out)           // (B*T, V)
{
  __shared__ float copy_v[VCHUNK];
  __shared__ float att[Sc];
  __shared__ float red_m[512];
  __shared__ float red_s[512];

  const int row = blockIdx.x;          // b*T + t
  const int b = row / Tc, t = row % Tc;
  const int tid = threadIdx.x;         // 0..511 == s index
  const float p = pgen[row];

  // ---- attention: mean over H, softmax over S, scale by (1-p) ----
  float a = 0.0f;
#pragma unroll
  for (int h = 0; h < Hc; ++h)
    a += __builtin_nontemporal_load(
        &attw[(((size_t)b * Hc + h) * Tc + t) * Sc + tid]);
  a *= (1.0f / Hc);

  red_m[tid] = a;
  __syncthreads();
  for (int off = 256; off > 0; off >>= 1) {
    if (tid < off) red_m[tid] = fmaxf(red_m[tid], red_m[tid + off]);
    __syncthreads();
  }
  const float amax = red_m[0];
  __syncthreads();
  const float e = expf(a - amax);
  red_s[tid] = e;
  __syncthreads();
  for (int off = 256; off > 0; off >>= 1) {
    if (tid < off) red_s[tid] += red_s[tid + off];
    __syncthreads();
  }
  const float asum = red_s[0];
  __syncthreads();
  att[tid] = (1.0f - p) * e / asum;

  // ---- online softmax statistics over the V=32000 row (b128 sweep) ----
  const v4f* __restrict__ frow4 = (const v4f*)(finalo + (size_t)row * Vc);
  float lm = -INFINITY, ls = 0.0f;
  for (int v4 = tid; v4 < Vc / 4; v4 += 512) {
    v4f x4 = frow4[v4];
#pragma unroll
    for (int i = 0; i < 4; ++i) {
      float x = x4[i];
      if (x > lm) { ls = ls * expf(lm - x) + 1.0f; lm = x; }
      else        { ls += expf(x - lm); }
    }
  }
  red_m[tid] = lm;
  red_s[tid] = ls;
  __syncthreads();
  for (int off = 256; off > 0; off >>= 1) {
    if (tid < off) {
      float m1 = red_m[tid],       s1 = red_s[tid];
      float m2 = red_m[tid + off], s2 = red_s[tid + off];
      float m  = fmaxf(m1, m2);
      red_m[tid] = m;
      red_s[tid] = s1 * expf(m1 - m) + s2 * expf(m2 - m);
    }
    __syncthreads();
  }
  const float gmax = red_m[0];
  const float ginv = 1.0f / red_s[0];
  const int   sidx = enc[(size_t)b * Sc + tid];
  v4f* __restrict__ orow4 = (v4f*)(out + (size_t)row * Vc);

  // ---- chunked scatter + emit (b128 in, NT b128 out) ----
  for (int c0 = 0; c0 < Vc; c0 += VCHUNK) {
    __syncthreads();
    for (int v = tid; v < VCHUNK; v += 512) copy_v[v] = 0.0f;
    __syncthreads();
    if (sidx >= c0 && sidx < c0 + VCHUNK)
      atomicAdd(&copy_v[sidx - c0], att[tid]);
    __syncthreads();
    const int v4lo = c0 / 4, v4hi = (c0 + VCHUNK) / 4;
    for (int v4 = v4lo + tid; v4 < v4hi; v4 += 512) {
      v4f x4 = frow4[v4];
      const float* cp = &copy_v[v4 * 4 - c0];
      v4f o4;
#pragma unroll
      for (int i = 0; i < 4; ++i)
        o4[i] = logf(p * expf(x4[i] - gmax) * ginv + cp[i] + 0.001f);
      __builtin_nontemporal_store(o4, &orow4[v4]);
    }
  }
}

// ---------------------------------------------------------------------------
extern "C" void kernel_launch(void* const* d_in, const int* in_sizes, int n_in,
                              void* d_out, int out_size, void* d_ws, size_t ws_size,
                              hipStream_t stream) {
  const float* dec   = (const float*)d_in[0];  // (B,T,D)
  const float* fin   = (const float*)d_in[1];  // (B,T,V)
  const float* attw  = (const float*)d_in[2];  // (B,H,T,S)
  const int*   enc   = (const int*)  d_in[3];  // (B,S)
  const float* Wv    = (const float*)d_in[4];  // (D,1)
  const float* bptr  = (const float*)d_in[5];  // (1,)
  float* pgen = (float*)d_ws;                  // (B*T) scratch
  float* outp = (float*)d_out;                 // (B,T,V)

  pgen_wmma_kernel<<<(Bc * Tc) / 16, 32, 0, stream>>>(dec, Wv, bptr, pgen);
  fused_row_kernel<<<Bc * Tc, 512, 0, stream>>>(fin, attw, enc, pgen, outp);
}
